// EGNN_71313636982933
// MI455X (gfx1250) — matrix operand order repaired and checked by reading, not compiled, not run on hardware
//
#include <hip/hip_runtime.h>
#include <hip/hip_bf16.h>

// ---------------------------------------------------------------------------
// EGNN forward for MI455X (gfx1250, wave32).
// All matmuls (K x 64 @ per-row tiles of 16) run on v_wmma_f32_16x16x32_bf16.
// Weights pre-packed to the wave32 WMMA B-fragment layout; activations staged
// in LDS as bf16; f32 accumulation; residual streams kept f32 in HBM;
// segment_sum via global_atomic_add_f32. Activations use hardware rcp/tanh
// transcendentals so the VALU epilogue can co-execute with the XDL WMMA pipe.
// ---------------------------------------------------------------------------

typedef __bf16 bf16;
typedef __attribute__((ext_vector_type(16))) __bf16 v16bf;
typedef __attribute__((ext_vector_type(8)))  __bf16 v8bf;
typedef __attribute__((ext_vector_type(8)))  float  v8f;

#define DEVINL __device__ __forceinline__

#ifndef __has_builtin
#define __has_builtin(x) 0
#endif

DEVINL float fast_rcp(float x) {
#if __has_builtin(__builtin_amdgcn_rcpf)
  return __builtin_amdgcn_rcpf(x);        // v_rcp_f32
#else
  return 1.0f / x;
#endif
}

DEVINL float siluf(float x) {             // x * sigmoid(x), no IEEE divide chain
  return x * fast_rcp(1.0f + __expf(-x)); // v_mul + v_exp + v_add + v_rcp + v_mul
}

DEVINL float fast_tanh(float x) {
#if __has_builtin(__builtin_amdgcn_tanhf)
  return __builtin_amdgcn_tanhf(x);       // v_tanh_f32 (gfx1250 TRANS op)
#elif __has_builtin(__builtin_amdgcn_tanh_f32)
  return __builtin_amdgcn_tanh_f32(x);
#else
  float e = __expf(2.0f * x);             // tanh = 1 - 2/(e^{2x}+1)
  return 1.0f - 2.0f * fast_rcp(e + 1.0f);
#endif
}

template<int ACT> DEVINL float actf(float v) {
  if (ACT == 0) return siluf(v);
  if (ACT == 1) return fast_tanh(v);
  return v;                                  // ACT == 2 : identity
}

DEVINL void lds_fence() { asm volatile("s_wait_dscnt 0" ::: "memory"); }

DEVINL void atomAddF(float* p, float v) { unsafeAtomicAdd(p, v); }

// A-fragment for 16x16x32 bf16 WMMA (wave32):
//   lanes 0-15  : row M=lane,    elems 0-7 = K(c+0..7),  8-15 = K(c+16..23)
//   lanes 16-31 : row M=lane-16, elems 0-7 = K(c+8..15), 8-15 = K(c+24..31)
DEVINL v16bf load_afrag(const bf16* base) {
  v8bf lo = *(const v8bf*)(base);
  v8bf hi = *(const v8bf*)(base + 16);
  v16bf a;
#pragma unroll
  for (int i = 0; i < 8; i++) { a[i] = lo[i]; a[8 + i] = hi[i]; }
  return a;
}

// Two-layer MLP on a 16-row tile:  Out[16,64] = act2( act1(A@B1 + b1) @ B2 + b2 )
// A: LDS bf16 tile, row stride sA, K = kChunks*32 (zero padded).
// B1/B2: packed B-fragments in global memory. Hid: LDS bf16 [16,64]. Out: LDS f32 [16,64].
template<int ACT1, int ACT2>
DEVINL void mlp2_tile(const bf16* __restrict__ A, int sA, int kChunks,
                      const bf16* __restrict__ B1, const float* __restrict__ b1,
                      const bf16* __restrict__ B2, const float* __restrict__ b2,
                      bf16* __restrict__ Hid, float* __restrict__ Out, int lane)
{
  const int row = lane & 15, hi = lane >> 4;
  __builtin_prefetch(B1, 0, 1);            // global_prefetch_b8 (weights -> WGP$/L2)
  __builtin_prefetch(B2, 0, 1);
  v8f acc[4] = {};
  for (int kc = 0; kc < kChunks; kc++) {
    v16bf a = load_afrag(A + row * sA + kc * 32 + hi * 8);
#pragma unroll
    for (int nb = 0; nb < 4; nb++) {
      v16bf b = *(const v16bf*)(B1 + (size_t)((kc * 4 + nb) * 32 + lane) * 16);
      acc[nb] = __builtin_amdgcn_wmma_f32_16x16x32_bf16(
          false, a, false, b, (short)0, acc[nb], false, false);
    }
  }
  // bias + act1, spill to LDS hidden tile as bf16.
  // C/D layout: vgpr r -> M = r + hi*8, N = nb*16 + (lane&15)
#pragma unroll
  for (int nb = 0; nb < 4; nb++)
#pragma unroll
    for (int r = 0; r < 8; r++) {
      int m = r + hi * 8, n = nb * 16 + row;
      Hid[m * 64 + n] = (bf16)actf<ACT1>(acc[nb][r] + b1[n]);
    }
  lds_fence();

  v8f acc2[4] = {};
#pragma unroll
  for (int kc = 0; kc < 2; kc++) {
    v16bf a = load_afrag(Hid + row * 64 + kc * 32 + hi * 8);
#pragma unroll
    for (int nb = 0; nb < 4; nb++) {
      v16bf b = *(const v16bf*)(B2 + (size_t)((kc * 4 + nb) * 32 + lane) * 16);
      acc2[nb] = __builtin_amdgcn_wmma_f32_16x16x32_bf16(
          false, a, false, b, (short)0, acc2[nb], false, false);
    }
  }
#pragma unroll
  for (int nb = 0; nb < 4; nb++)
#pragma unroll
    for (int r = 0; r < 8; r++) {
      int m = r + hi * 8, n = nb * 16 + row;
      Out[m * 64 + n] = actf<ACT2>(acc2[nb][r] + b2[n]);
    }
  lds_fence();
}

// ---------------------------------------------------------------------------
// Weight packing: W[K,64] f32 -> bf16 B-fragments, K zero-padded to Kpad.
// dst[((kc*4+nb)*32 + lane)*16 + e] = W[kc*32 + (lane>=16?8:0) + (e<8 ? e : 8+e)][nb*16 + lane%16]
// swap12: remap packed rows 64..191 (swap two 64-row segments) so the staged
// activation layout can keep both MLP inputs K-contiguous.
// ---------------------------------------------------------------------------
__global__ void pack_b_kernel(const float* __restrict__ src, bf16* __restrict__ dst,
                              int K, int Kpad, int swap12)
{
  int idx = blockIdx.x * blockDim.x + threadIdx.x;
  if (idx >= Kpad * 64) return;
  int e  = idx & 15;
  int ln = (idx >> 4) & 31;
  int nb = (idx >> 9) & 3;
  int kc = idx >> 11;
  int n  = nb * 16 + (ln & 15);
  int k  = kc * 32 + ((ln >> 4) ? 8 : 0) + (e < 8 ? e : 8 + e);
  int ks = k;
  if (swap12 && k >= 64 && k < 192) ks = (k < 128) ? (k + 64) : (k - 64);
  float v = (ks < K) ? src[(size_t)ks * 64 + n] : 0.0f;
  dst[idx] = (bf16)v;
}

__global__ void zero_kernel(float* __restrict__ p, int n) {
  int i = blockIdx.x * blockDim.x + threadIdx.x;
  if (i < n) p[i] = 0.0f;
}

__global__ void dist_kernel(const float* __restrict__ pos,
                            const long long* __restrict__ send,
                            const long long* __restrict__ rec,
                            bf16* __restrict__ distb, int E)
{
  int e = blockIdx.x * blockDim.x + threadIdx.x;
  if (e >= E) return;
  size_t s = (size_t)send[e] * 3, r = (size_t)rec[e] * 3;
  float dx = pos[s] - pos[r], dy = pos[s + 1] - pos[r + 1], dz = pos[s + 2] - pos[r + 2];
  distb[e] = (bf16)sqrtf(dx * dx + dy * dy + dz * dz);
}

// ---------------------------------------------------------------------------
// Embed: h = Lin(silu(Lin(x))), pe = Lin(silu(Lin(rw))).  K pad 11->32, 15->32.
// ---------------------------------------------------------------------------
__global__ __launch_bounds__(128)
void embed_kernel(const float* __restrict__ x, const float* __restrict__ rw,
                  float* __restrict__ h, float* __restrict__ pe,
                  const bf16* We1p, const float* be1, const bf16* We2p, const float* be2,
                  const bf16* Wp1p, const float* bp1, const bf16* Wp2p, const float* bp2,
                  int nTiles)
{
  __shared__ __align__(16) bf16  sState[4][16 * 32];
  __shared__ __align__(16) bf16  sHid[4][16 * 64];
  __shared__ __align__(16) float sOut[4][16 * 64];
  const int wave = threadIdx.x >> 5, lane = threadIdx.x & 31;
  const int t = blockIdx.x * 4 + wave;
  if (t >= nTiles) return;
  const int n0 = t * 16;
  bf16* st = sState[wave];
  const int m2 = lane >> 1, half = lane & 1;

  for (int i = lane; i < 512; i += 32) {
    int m = i >> 5, c = i & 31;
    st[i] = (c < 11) ? (bf16)x[(size_t)(n0 + m) * 11 + c] : (bf16)0.0f;
  }
  lds_fence();
  mlp2_tile<0, 2>(st, 32, 1, We1p, be1, We2p, be2, sHid[wave], sOut[wave], lane);
  {
    float* dst = h + (size_t)(n0 + m2) * 64 + half * 32;
    const float* s2 = sOut[wave] + m2 * 64 + half * 32;
#pragma unroll
    for (int j = 0; j < 32; j++) dst[j] = s2[j];
  }
  for (int i = lane; i < 512; i += 32) {
    int m = i >> 5, c = i & 31;
    st[i] = (c < 15) ? (bf16)rw[(size_t)(n0 + m) * 15 + c] : (bf16)0.0f;
  }
  lds_fence();
  mlp2_tile<0, 2>(st, 32, 1, Wp1p, bp1, Wp2p, bp2, sHid[wave], sOut[wave], lane);
  {
    float* dst = pe + (size_t)(n0 + m2) * 64 + half * 32;
    const float* s2 = sOut[wave] + m2 * 64 + half * 32;
#pragma unroll
    for (int j = 0; j < 32; j++) dst[j] = s2[j];
  }
}

// ---------------------------------------------------------------------------
// Edge kernel: state tile layout [hs|hr|ps|pr|dist|0pad] (stride 288).
//   msg     : K window [0..288)  with row-swapped Wm1 (silu, silu) -> aggr
//   msg_pos : K window [128..288) = [ps|pr|dist|0] identity Wq1 (tanh, tanh) -> aggr_pos
// ---------------------------------------------------------------------------
__global__ __launch_bounds__(128)
void edge_kernel(const float* __restrict__ h, const float* __restrict__ pe,
                 const bf16* __restrict__ distb,
                 const long long* __restrict__ send, const long long* __restrict__ rec,
                 const bf16* Wm1p, const float* bm1, const bf16* Wm2p, const float* bm2,
                 const bf16* Wq1p, const float* bq1, const bf16* Wq2p, const float* bq2,
                 float* __restrict__ aggr, float* __restrict__ aggrp, int nTiles)
{
  __shared__ __align__(16) bf16  sState[4][16 * 288];
  __shared__ __align__(16) bf16  sHid[4][16 * 64];
  __shared__ __align__(16) float sOut[4][16 * 64];
  __shared__ int sIdx[4][32];
  const int wave = threadIdx.x >> 5, lane = threadIdx.x & 31;
  const int t = blockIdx.x * 4 + wave;
  if (t >= nTiles) return;
  const int e0 = t * 16;

  if (lane < 16) {
    sIdx[wave][lane]      = (int)send[e0 + lane];
    sIdx[wave][16 + lane] = (int)rec[e0 + lane];
  }
  lds_fence();

  bf16* st = sState[wave];
  const int c2 = lane * 2;
  for (int m = 0; m < 16; m++) {
    int s = sIdx[wave][m], r = sIdx[wave][16 + m];
    float2 a = *(const float2*)(h  + (size_t)s * 64 + c2);
    float2 b = *(const float2*)(h  + (size_t)r * 64 + c2);
    float2 c = *(const float2*)(pe + (size_t)s * 64 + c2);
    float2 d = *(const float2*)(pe + (size_t)r * 64 + c2);
    st[m * 288 +       c2] = (bf16)a.x; st[m * 288 +   1 + c2] = (bf16)a.y;
    st[m * 288 +  64 + c2] = (bf16)b.x; st[m * 288 +  65 + c2] = (bf16)b.y;
    st[m * 288 + 128 + c2] = (bf16)c.x; st[m * 288 + 129 + c2] = (bf16)c.y;
    st[m * 288 + 192 + c2] = (bf16)d.x; st[m * 288 + 193 + c2] = (bf16)d.y;
  }
  if (lane < 16) st[lane * 288 + 256] = distb[e0 + lane];
  {
    const int row = lane & 15, hi = lane >> 4;
    int base = row * 288 + 257 + hi * 16;
    int cnt  = hi ? 15 : 16;
    for (int i = 0; i < cnt; i++) st[base + i] = (bf16)0.0f;
  }
  lds_fence();

  mlp2_tile<0, 0>(st, 288, 9, Wm1p, bm1, Wm2p, bm2, sHid[wave], sOut[wave], lane);
  {
    const int m = lane >> 1, half = lane & 1;
    const int r = sIdx[wave][16 + m];
    float* dst = aggr + (size_t)r * 64 + half * 32;
    const float* s2 = sOut[wave] + m * 64 + half * 32;
#pragma unroll
    for (int j = 0; j < 32; j++) atomAddF(dst + j, s2[j]);
  }
  mlp2_tile<1, 1>(st + 128, 288, 5, Wq1p, bq1, Wq2p, bq2, sHid[wave], sOut[wave], lane);
  {
    const int m = lane >> 1, half = lane & 1;
    const int r = sIdx[wave][16 + m];
    float* dst = aggrp + (size_t)r * 64 + half * 32;
    const float* s2 = sOut[wave] + m * 64 + half * 32;
#pragma unroll
    for (int j = 0; j < 32; j++) atomAddF(dst + j, s2[j]);
  }
}

// ---------------------------------------------------------------------------
// Node update: state tile [h|aggr|pe|aggr_pos] (stride 256).
//   upd    : K [0..192) with row-swapped Wu1 (silu, none) ; h += upd
//   upd_pe : K [128..256) = [pe|aggr_pos] identity Wv1 (tanh, tanh) ; pe += upd_pe
// ---------------------------------------------------------------------------
__global__ __launch_bounds__(128)
void node_kernel(float* __restrict__ h, float* __restrict__ pe,
                 const float* __restrict__ aggr, const float* __restrict__ aggrp,
                 const bf16* Wu1p, const float* bu1, const bf16* Wu2p, const float* bu2,
                 const bf16* Wv1p, const float* bv1, const bf16* Wv2p, const float* bv2,
                 int nTiles)
{
  __shared__ __align__(16) bf16  sState[4][16 * 256];
  __shared__ __align__(16) bf16  sHid[4][16 * 64];
  __shared__ __align__(16) float sOut[4][16 * 64];
  const int wave = threadIdx.x >> 5, lane = threadIdx.x & 31;
  const int t = blockIdx.x * 4 + wave;
  if (t >= nTiles) return;
  const int n0 = t * 16;
  bf16* st = sState[wave];
  const int c2 = lane * 2;
  for (int m = 0; m < 16; m++) {
    size_t ro = (size_t)(n0 + m) * 64 + c2;
    float2 a = *(const float2*)(h     + ro);
    float2 b = *(const float2*)(aggr  + ro);
    float2 c = *(const float2*)(pe    + ro);
    float2 d = *(const float2*)(aggrp + ro);
    st[m * 256 +       c2] = (bf16)a.x; st[m * 256 +   1 + c2] = (bf16)a.y;
    st[m * 256 +  64 + c2] = (bf16)b.x; st[m * 256 +  65 + c2] = (bf16)b.y;
    st[m * 256 + 128 + c2] = (bf16)c.x; st[m * 256 + 129 + c2] = (bf16)c.y;
    st[m * 256 + 192 + c2] = (bf16)d.x; st[m * 256 + 193 + c2] = (bf16)d.y;
  }
  lds_fence();

  mlp2_tile<0, 2>(st, 256, 6, Wu1p, bu1, Wu2p, bu2, sHid[wave], sOut[wave], lane);
  {
    const int m = lane >> 1, half = lane & 1;
    float* dst = h + (size_t)(n0 + m) * 64 + half * 32;
    const float* s2 = sOut[wave] + m * 64 + half * 32;
#pragma unroll
    for (int j = 0; j < 32; j++) dst[j] += s2[j];
  }
  mlp2_tile<1, 1>(st + 128, 256, 4, Wv1p, bv1, Wv2p, bv2, sHid[wave], sOut[wave], lane);
  {
    const int m = lane >> 1, half = lane & 1;
    float* dst = pe + (size_t)(n0 + m) * 64 + half * 32;
    const float* s2 = sOut[wave] + m * 64 + half * 32;
#pragma unroll
    for (int j = 0; j < 32; j++) dst[j] += s2[j];     // pe = upd_pe + pe
  }
}

// ---------------------------------------------------------------------------
// Readout: per-node MLP then global_add_pool by `batch` via atomics.
// ---------------------------------------------------------------------------
__global__ __launch_bounds__(128)
void readout_kernel(const float* __restrict__ h, const long long* __restrict__ batch,
                    const bf16* Wr1p, const float* br1, const bf16* Wr2p, const float* br2,
                    float* __restrict__ g, int nTiles)
{
  __shared__ __align__(16) bf16  sState[4][16 * 64];
  __shared__ __align__(16) bf16  sHid[4][16 * 64];
  __shared__ __align__(16) float sOut[4][16 * 64];
  const int wave = threadIdx.x >> 5, lane = threadIdx.x & 31;
  const int t = blockIdx.x * 4 + wave;
  if (t >= nTiles) return;
  const int n0 = t * 16;
  bf16* st = sState[wave];
  const int c2 = lane * 2;
  for (int m = 0; m < 16; m++) {
    float2 a = *(const float2*)(h + (size_t)(n0 + m) * 64 + c2);
    st[m * 64 + c2] = (bf16)a.x; st[m * 64 + 1 + c2] = (bf16)a.y;
  }
  lds_fence();
  mlp2_tile<0, 2>(st, 64, 2, Wr1p, br1, Wr2p, br2, sHid[wave], sOut[wave], lane);
  {
    const int m = lane >> 1, half = lane & 1;
    const int b = (int)batch[n0 + m];
    float* dst = g + (size_t)b * 64 + half * 32;
    const float* s2 = sOut[wave] + m * 64 + half * 32;
#pragma unroll
    for (int j = 0; j < 32; j++) atomAddF(dst + j, s2[j]);
  }
}

// Final tiny MLP: out[gi] = silu(g @ Wo1 + bo1) @ Wo2 + bo2.  ~2 MFLOP -> VALU.
__global__ void final_kernel(const float* __restrict__ g,
                             const float* __restrict__ Wo1, const float* __restrict__ bo1,
                             const float* __restrict__ Wo2, const float* __restrict__ bo2,
                             float* __restrict__ out)
{
  int t = blockIdx.x * blockDim.x + threadIdx.x;     // 256 graphs
  const float* gr = g + (size_t)t * 64;
  float accum = 0.0f;
  for (int j = 0; j < 64; j++) {
    float s = bo1[j];
    for (int k = 0; k < 64; k++) s += gr[k] * Wo1[(size_t)k * 64 + j];
    accum += siluf(s) * Wo2[j];
  }
  out[t] = accum + bo2[0];
}

// ---------------------------------------------------------------------------
extern "C" void kernel_launch(void* const* d_in, const int* in_sizes, int n_in,
                              void* d_out, int out_size, void* d_ws, size_t ws_size,
                              hipStream_t stream)
{
  (void)in_sizes; (void)n_in; (void)out_size; (void)ws_size;
  const int N = 50000, E = 800000, H = 64, L = 4, NG = 256;

  const float*     x     = (const float*)d_in[0];
  const float*     pos   = (const float*)d_in[1];
  const float*     rw    = (const float*)d_in[2];
  const long long* ei    = (const long long*)d_in[3];
  const long long* batch = (const long long*)d_in[4];
  const long long* send  = ei;
  const long long* rec   = ei + E;
  const float *We1 = (const float*)d_in[5],  *be1 = (const float*)d_in[6];
  const float *We2 = (const float*)d_in[7],  *be2 = (const float*)d_in[8];
  const float *Wp1 = (const float*)d_in[9],  *bp1 = (const float*)d_in[10];
  const float *Wp2 = (const float*)d_in[11], *bp2 = (const float*)d_in[12];
  const float *Wm1 = (const float*)d_in[13], *bm1 = (const float*)d_in[14];
  const float *Wm2 = (const float*)d_in[15], *bm2 = (const float*)d_in[16];
  const float *Wq1 = (const float*)d_in[17], *bq1 = (const float*)d_in[18];
  const float *Wq2 = (const float*)d_in[19], *bq2 = (const float*)d_in[20];
  const float *Wu1 = (const float*)d_in[21], *bu1 = (const float*)d_in[22];
  const float *Wu2 = (const float*)d_in[23], *bu2 = (const float*)d_in[24];
  const float *Wv1 = (const float*)d_in[25], *bv1 = (const float*)d_in[26];
  const float *Wv2 = (const float*)d_in[27], *bv2 = (const float*)d_in[28];
  const float *Wr1 = (const float*)d_in[29], *br1 = (const float*)d_in[30];
  const float *Wr2 = (const float*)d_in[31], *br2 = (const float*)d_in[32];
  const float *Wo1 = (const float*)d_in[33], *bo1 = (const float*)d_in[34];
  const float *Wo2 = (const float*)d_in[35], *bo2 = (const float*)d_in[36];
  float* out = (float*)d_out;

  // Workspace bump allocator (256B aligned).  Total ~54 MB.
  char* base = (char*)d_ws; size_t off = 0;
  auto alloc = [&](size_t bytes) -> char* {
    char* p = base + off; off += (bytes + 255) & ~(size_t)255; return p;
  };
  float* h     = (float*)alloc((size_t)N * H * 4);
  float* pe    = (float*)alloc((size_t)N * H * 4);
  float* aggr  = (float*)alloc((size_t)N * H * 4);
  float* aggrp = (float*)alloc((size_t)N * H * 4);
  bf16*  distb = (bf16*)alloc((size_t)E * 2);
  float* g     = (float*)alloc((size_t)NG * H * 4);

  auto pk = [&](const float* src, int K, int Kpad, int swap12) -> bf16* {
    bf16* dst = (bf16*)alloc((size_t)Kpad * 64 * 2);
    int total = Kpad * 64;
    pack_b_kernel<<<(total + 255) / 256, 256, 0, stream>>>(src, dst, K, Kpad, swap12);
    return dst;
  };
  bf16* We1p = pk(We1, 11, 32, 0);
  bf16* We2p = pk(We2, 64, 64, 0);
  bf16* Wp1p = pk(Wp1, 15, 32, 0);
  bf16* Wp2p = pk(Wp2, 64, 64, 0);
  bf16* Wr1p = pk(Wr1, 64, 64, 0);
  bf16* Wr2p = pk(Wr2, 64, 64, 0);
  bf16 *Wm1p[4], *Wm2p[4], *Wq1p[4], *Wq2p[4], *Wu1p[4], *Wu2p[4], *Wv1p[4], *Wv2p[4];
  for (int l = 0; l < L; l++) {
    Wm1p[l] = pk(Wm1 + (size_t)l * 257 * 64, 257, 288, 1);  // rows permuted: [hs|hr|ps|pr|dist]
    Wm2p[l] = pk(Wm2 + (size_t)l * 64 * 64,   64,  64, 0);
    Wq1p[l] = pk(Wq1 + (size_t)l * 129 * 64, 129, 160, 0);  // [ps|pr|dist] identity
    Wq2p[l] = pk(Wq2 + (size_t)l * 64 * 64,   64,  64, 0);
    Wu1p[l] = pk(Wu1 + (size_t)l * 192 * 64, 192, 192, 1);  // rows permuted: [h|aggr|pe]
    Wu2p[l] = pk(Wu2 + (size_t)l * 64 * 64,   64,  64, 0);
    Wv1p[l] = pk(Wv1 + (size_t)l * 128 * 64, 128, 128, 0);  // [pe|aggr_pos] identity
    Wv2p[l] = pk(Wv2 + (size_t)l * 64 * 64,   64,  64, 0);
  }

  dist_kernel<<<(E + 255) / 256, 256, 0, stream>>>(pos, send, rec, distb, E);

  const int nodeTiles  = N / 16;                 // 3125
  const int nodeBlocks = (nodeTiles + 3) / 4;
  const int edgeTiles  = E / 16;                 // 50000
  const int edgeBlocks = (edgeTiles + 3) / 4;

  embed_kernel<<<nodeBlocks, 128, 0, stream>>>(x, rw, h, pe,
      We1p, be1, We2p, be2, Wp1p, bp1, Wp2p, bp2, nodeTiles);

  for (int l = 0; l < L; l++) {
    zero_kernel<<<(N * H + 255) / 256, 256, 0, stream>>>(aggr,  N * H);
    zero_kernel<<<(N * H + 255) / 256, 256, 0, stream>>>(aggrp, N * H);
    edge_kernel<<<edgeBlocks, 128, 0, stream>>>(h, pe, distb, send, rec,
        Wm1p[l], bm1 + l * H, Wm2p[l], bm2 + l * H,
        Wq1p[l], bq1 + l * H, Wq2p[l], bq2 + l * H,
        aggr, aggrp, edgeTiles);
    node_kernel<<<nodeBlocks, 128, 0, stream>>>(h, pe, aggr, aggrp,
        Wu1p[l], bu1 + l * H, Wu2p[l], bu2 + l * H,
        Wv1p[l], bv1 + l * H, Wv2p[l], bv2 + l * H, nodeTiles);
  }

  zero_kernel<<<(NG * H + 255) / 256, 256, 0, stream>>>(g, NG * H);
  readout_kernel<<<nodeBlocks, 128, 0, stream>>>(h, batch,
      Wr1p, br1, Wr2p, br2, g, nodeTiles);
  final_kernel<<<1, 256, 0, stream>>>(g, Wo1, bo1, Wo2, bo2, out);
}
